// FlashAttention_50843822850151
// MI455X (gfx1250) — matrix operand order repaired and checked by reading, not compiled, not run on hardware
//
#include <hip/hip_runtime.h>

// ---------------------------------------------------------------------------
// MI455X (gfx1250) fused attention.
//  * all matrix math through v_wmma_f32_16x16x32_f16 (high-rate matrix path)
//  * GEMMs: 128x128 block tile, double-buffered LDS, 64 WMMA / block / K-step
//  * attention: TDM (tensor_load_to_lds) stages Q and K blocks into LDS,
//    s_wait_tensorcnt + barrier publishes; 256KB of the WGP's 320KB LDS
// ---------------------------------------------------------------------------

typedef __attribute__((ext_vector_type(16))) _Float16 v16h;
typedef __attribute__((ext_vector_type(8)))  float    v8f;
typedef __attribute__((ext_vector_type(4)))  unsigned int u32x4;
typedef __attribute__((ext_vector_type(8)))  int      i32x8;
typedef __attribute__((ext_vector_type(4)))  int      i32x4;

#define B_    2
#define S_    2048
#define H_    2048
#define NH_   16
#define HD_   128
#define KBS_  256          // reference key-block size (per-block softmax width)
#define QT_   128          // queries per attention workgroup
#define SCALE_ 0.08838834764831845f   // 1/sqrt(128)

#if __has_builtin(__builtin_amdgcn_tensor_load_to_lds)
#define HAVE_TDM 1
#else
#define HAVE_TDM 0
#endif

// ---- WMMA fragment loaders (ISA 7.12.2 layouts, wave32) --------------------
// A 16x32 f16: row m = lane&15; lane<16 -> K {k0..k0+7, k0+16..k0+23},
//              lane>=16 -> K {k0+8..k0+15, k0+24..k0+31}
__device__ __forceinline__ v16h load_frag_a(const _Float16* t, int ld, int m0, int k0) {
  const int lane = threadIdx.x & 31;
  const _Float16* p = t + (size_t)(m0 + (lane & 15)) * ld + k0 + ((lane & 16) ? 8 : 0);
  v16h f;
#pragma unroll
  for (int i = 0; i < 8; ++i) { f[i] = p[i]; f[i + 8] = p[i + 16]; }
  return f;
}
// B 32x16 f16 (stored [N][K] row-major): col n = lane&15;
// lane<16 -> K {k0..k0+7, k0+16..k0+23}, lane>=16 -> K {k0+8..k0+15, k0+24..k0+31}
__device__ __forceinline__ v16h load_frag_b(const _Float16* t, int ld, int n0, int k0) {
  const int lane = threadIdx.x & 31;
  const _Float16* p = t + (size_t)(n0 + (lane & 15)) * ld + k0 + ((lane & 16) ? 8 : 0);
  v16h f;
#pragma unroll
  for (int i = 0; i < 8; ++i) { f[i] = p[i]; f[i + 8] = p[i + 16]; }
  return f;
}

// ---- TDM: flat contiguous copy of n f16 elements, global -> LDS ------------
// 1-D D# descriptor: count=1, data_size=2B, tile_dim0=n, tile_dim1=1, type=2.
// Issue from one wave; producer does s_wait_tensorcnt 0; barrier publishes.
#if HAVE_TDM
__device__ __forceinline__ void tdm_load_flat(unsigned lds_off, const void* g, unsigned n) {
  const unsigned long long ga = (unsigned long long)g;
  u32x4 g0;
  g0[0] = 1u;                                             // count=1 (valid user D#)
  g0[1] = lds_off;                                        // lds_addr
  g0[2] = (unsigned)(ga & 0xFFFFFFFFu);                   // global_addr[31:0]
  g0[3] = (unsigned)((ga >> 32) & 0x01FFFFFFu) | (2u << 30); // global_addr[56:32] | type=2
  i32x8 g1;
  g1[0] = 0x00010000;                                     // data_size=1 (2 bytes)
  g1[1] = (int)((n & 0xFFFFu) << 16);                     // tensor_dim0[15:0]
  g1[2] = (int)(((n >> 16) & 0xFFFFu) | (1u << 16));      // tensor_dim0[31:16], tensor_dim1=1
  g1[3] = (int)((n & 0xFFFFu) << 16);                     // tile_dim0 = n
  g1[4] = 1;                                              // tile_dim1 = 1
  g1[5] = (int)n;                                         // tensor_dim0_stride
  g1[6] = 0; g1[7] = 0;
  const i32x4 gz = {0, 0, 0, 0};
#if __clang_major__ >= 23
  const i32x8 gz8 = {0, 0, 0, 0, 0, 0, 0, 0};
  __builtin_amdgcn_tensor_load_to_lds(g0, g1, gz, gz, gz8, 0);
#else
  __builtin_amdgcn_tensor_load_to_lds(g0, g1, gz, gz, 0);
#endif
}
#endif

// ---- fp32 -> f16 conversion ------------------------------------------------
__global__ __launch_bounds__(256) void cvt_f32_f16(const float* __restrict__ x,
                                                   _Float16* __restrict__ y, int n4) {
  int i = blockIdx.x * 256 + threadIdx.x;
  if (i < n4) {
    float4 v = ((const float4*)x)[i];
    y[i * 4 + 0] = (_Float16)v.x;
    y[i * 4 + 1] = (_Float16)v.y;
    y[i * 4 + 2] = (_Float16)v.z;
    y[i * 4 + 3] = (_Float16)v.w;
  }
}

// ---- C = A(row-major [M,K]) * W(row-major [N,K])^T -------------------------
// 128x128 block tile, 8 waves each doing 32x64 (8 WMMA / K-step), LDS
// double-buffered so next tile's staging overlaps the WMMA burst.
// MODE 0: write f16 head-split [B,NH,S,HD]   (QKV projections)
// MODE 1: write f32 row-major  [M,N]         (output projection)
template <int MODE>
__global__ __launch_bounds__(256) void gemm_xwT(const _Float16* __restrict__ A,
                                                const _Float16* __restrict__ W,
                                                void* __restrict__ dstv,
                                                int M, int N, int K) {
  __shared__ _Float16 sA[2][128 * 32];
  __shared__ _Float16 sB[2][128 * 32];
  const int tid = threadIdx.x, lane = tid & 31, wv = tid >> 5;
  const int tm = blockIdx.x * 128, tn = blockIdx.y * 128;
  const int wm = (wv & 3) * 32, wn = (wv >> 2) * 64;
  const v8f vzero = {0.f, 0.f, 0.f, 0.f, 0.f, 0.f, 0.f, 0.f};
  v8f acc[2][4];
#pragma unroll
  for (int i = 0; i < 2; ++i)
#pragma unroll
    for (int j = 0; j < 4; ++j) acc[i][j] = vzero;

  // stage one 128x32 tile of A and W into LDS buffer `buf` (2 uint4 per thread each)
  auto stage = [&](int buf, int k0) {
#pragma unroll
    for (int p = 0; p < 2; ++p) {
      const int ee = tid * 8 + p * 2048, rr = ee >> 5, cc = ee & 31;
      *(uint4*)&sA[buf][ee] = *(const uint4*)&A[(size_t)(tm + rr) * K + k0 + cc];
      *(uint4*)&sB[buf][ee] = *(const uint4*)&W[(size_t)(tn + rr) * K + k0 + cc];
      if (k0 + 32 < K) {  // global_prefetch_b8 for the tile after next
        __builtin_prefetch(&A[(size_t)(tm + rr) * K + k0 + 32 + cc], 0, 1);
        __builtin_prefetch(&W[(size_t)(tn + rr) * K + k0 + 32 + cc], 0, 1);
      }
    }
  };

  const int nIter = K / 32;
  stage(0, 0);
  for (int i = 0; i < nIter; ++i) {
    __syncthreads();                       // buf (i&1) ready for everyone
    const int cur = i & 1;
    if (i + 1 < nIter) stage(cur ^ 1, (i + 1) * 32);   // overlaps WMMAs below
    const _Float16* cA = sA[cur];
    const _Float16* cB = sB[cur];
    v16h a0 = load_frag_a(cA, 32, wm, 0);
    v16h a1 = load_frag_a(cA, 32, wm + 16, 0);
#pragma unroll
    for (int nt = 0; nt < 4; ++nt) {
      v16h b = load_frag_b(cB, 32, wn + nt * 16, 0);
      acc[0][nt] = __builtin_amdgcn_wmma_f32_16x16x32_f16(false, a0, false, b, (short)0, acc[0][nt], false, false);
      acc[1][nt] = __builtin_amdgcn_wmma_f32_16x16x32_f16(false, a1, false, b, (short)0, acc[1][nt], false, false);
    }
  }

  const int hi = (lane >> 4) & 1, col = lane & 15;
#pragma unroll
  for (int mi = 0; mi < 2; ++mi) {
#pragma unroll
    for (int nt = 0; nt < 4; ++nt) {
#pragma unroll
      for (int r = 0; r < 8; ++r) {
        const float v = acc[mi][nt][r];
        const int m = tm + wm + mi * 16 + hi * 8 + r;
        const int n = tn + wn + nt * 16 + col;
        if (MODE == 0) {
          _Float16* dst = (_Float16*)dstv;
          const int b = m >> 11, s = m & (S_ - 1);
          const int h = n >> 7,  d = n & (HD_ - 1);
          dst[(((size_t)b * NH_ + h) * S_ + s) * HD_ + d] = (_Float16)v;
        } else {
          float* dst = (float*)dstv;
          dst[(size_t)m * N + n] = v;
        }
      }
    }
  }
}

// ---- attention: per-(b,h), 128 queries per workgroup, 8 waves --------------
// LDS: scores 8*16*256 f32 (128KB, also reused to stage Q at start)
//      kv     256*128 f16  (64KB, K then V^T)
//      P      8*16*256 f16 (64KB)
#define SM_SCORE 0
#define SM_KV    (131072)
#define SM_P     (131072 + 65536)
#define SM_TOTAL (262144)

__global__ __launch_bounds__(256) void attn_kernel(const _Float16* __restrict__ Q,
                                                   const _Float16* __restrict__ Kg,
                                                   const _Float16* __restrict__ Vg,
                                                   _Float16* __restrict__ Mrg) {
  extern __shared__ char smem[];
  float*    sScore = (float*)(smem + SM_SCORE);
  _Float16* sKV    = (_Float16*)(smem + SM_KV);
  _Float16* sP     = (_Float16*)(smem + SM_P);
  _Float16* sQ     = (_Float16*)(smem + SM_SCORE);   // alias: Q staging before scores

  const int tid = threadIdx.x, lane = tid & 31, wv = tid >> 5;
  const int hi = (lane >> 4) & 1, col = lane & 15;
  const int bh = blockIdx.y;                   // b*NH + h
  const int q0 = blockIdx.x * QT_;             // first query of this block
  const int bi = q0 >> 8;                      // reference 256-block index
  const size_t headOff = (size_t)bh * S_ * HD_;

  // ---- stage Q rows [q0, q0+128) (contiguous 32KB) via TDM ----
#if HAVE_TDM
  if (wv == 0) {
    tdm_load_flat(SM_SCORE, Q + headOff + (size_t)q0 * HD_, QT_ * HD_);
    __builtin_amdgcn_s_wait_tensorcnt(0);
  }
#else
  for (int e = tid * 8; e < QT_ * HD_; e += 256 * 8)
    *(uint4*)&sQ[e] = *(const uint4*)&Q[headOff + (size_t)q0 * HD_ + e];
#endif
  __syncthreads();

  // per-wave persistent Q fragments: 16 rows x 128 K  (4 k-tiles of 32)
  v16h qf[4];
#pragma unroll
  for (int kk = 0; kk < 4; ++kk) qf[kk] = load_frag_a(sQ, HD_, wv * 16, kk * 32);
  __syncthreads();   // sQ region now free for scores

  const v8f vzero = {0.f, 0.f, 0.f, 0.f, 0.f, 0.f, 0.f, 0.f};
  v8f o[8];
#pragma unroll
  for (int i = 0; i < 8; ++i) o[i] = vzero;

  float*    myS = sScore + (size_t)wv * 16 * 256;
  _Float16* myP = sP     + (size_t)wv * 16 * 256;
  const int qrow_base = q0 + wv * 16 + hi * 8;

  for (int j = 0; j <= bi; ++j) {
    // ---- stage K block j (256x128 f16, contiguous 64KB) via TDM ----
    const _Float16* kg = Kg + headOff + (size_t)j * KBS_ * HD_;
#if HAVE_TDM
    if (wv == 0) {
      tdm_load_flat(SM_KV, kg, KBS_ * HD_);
      __builtin_amdgcn_s_wait_tensorcnt(0);
    }
#else
    for (int e = tid * 8; e < KBS_ * HD_; e += 256 * 8)
      *(uint4*)&sKV[e] = *(const uint4*)&kg[e];
#endif
    __syncthreads();

    // ---- scores: this wave's 16 queries vs all 256 keys ----
    for (int t = 0; t < 16; ++t) {
      v8f s = vzero;
#pragma unroll
      for (int kk = 0; kk < 4; ++kk) {
        v16h bf = load_frag_b(sKV, HD_, t * 16, kk * 32);
        s = __builtin_amdgcn_wmma_f32_16x16x32_f16(false, qf[kk], false, bf, (short)0, s, false, false);
      }
      const int key = j * KBS_ + t * 16 + col;
#pragma unroll
      for (int r = 0; r < 8; ++r) {
        float v = s[r] * SCALE_;
        if (j == bi && key > qrow_base + r) v = -1e30f;   // causal, diagonal block only
        myS[(hi * 8 + r) * 256 + t * 16 + col] = v;
      }
    }

    // ---- per-block softmax (independent per key block, per query row) ----
    for (int row = 0; row < 16; ++row) {
      const float* rp = myS + row * 256;
      float x[8], mx = -1e30f;
#pragma unroll
      for (int u = 0; u < 8; ++u) { x[u] = rp[lane + u * 32]; mx = fmaxf(mx, x[u]); }
#pragma unroll
      for (int off = 16; off >= 1; off >>= 1) mx = fmaxf(mx, __shfl_xor(mx, off, 32));
      float sum = 0.f;
#pragma unroll
      for (int u = 0; u < 8; ++u) { x[u] = __expf(x[u] - mx); sum += x[u]; }
#pragma unroll
      for (int off = 16; off >= 1; off >>= 1) sum += __shfl_xor(sum, off, 32);
      const float inv = 1.f / sum;   // sum >= 1 (diagonal key always unmasked)
#pragma unroll
      for (int u = 0; u < 8; ++u) myP[row * 256 + lane + u * 32] = (_Float16)(x[u] * inv);
    }
    __syncthreads();   // all waves done with K tile

    // ---- stage V^T (128 x 256) into same LDS buffer (transpose: manual) ----
    const _Float16* vg = Vg + headOff + (size_t)j * KBS_ * HD_;
    for (int e = tid * 4; e < KBS_ * HD_; e += 256 * 4) {
      const int key = e >> 7, d = e & 127;
      uint2 raw = *(const uint2*)&vg[e];
      const _Float16* tp = (const _Float16*)&raw;
#pragma unroll
      for (int u = 0; u < 4; ++u) sKV[(size_t)(d + u) * 256 + key] = tp[u];
    }
    __syncthreads();

    // ---- O += P @ V  (16x256 f16 * 256x128 f16 per wave) ----
    for (int kt = 0; kt < 8; ++kt) {
      v16h pa = load_frag_a(myP, 256, 0, kt * 32);
#pragma unroll
      for (int nt = 0; nt < 8; ++nt) {
        v16h vb = load_frag_b(sKV, 256, nt * 16, kt * 32);
        o[nt] = __builtin_amdgcn_wmma_f32_16x16x32_f16(false, pa, false, vb, (short)0, o[nt], false, false);
      }
    }
    __syncthreads();   // before next j overwrites sKV
  }

  // reference norm = sum of per-block softmax weights = (bi+1) exactly
  const float invn = 1.f / ((float)(bi + 1) + 1e-6f);
  const int b = bh >> 4, h = bh & 15;
#pragma unroll
  for (int nt = 0; nt < 8; ++nt) {
#pragma unroll
    for (int r = 0; r < 8; ++r) {
      const int q = q0 + wv * 16 + hi * 8 + r;
      const int d = nt * 16 + col;
      Mrg[(((size_t)b * S_ + q) * NH_ + h) * HD_ + d] = (_Float16)(o[nt][r] * invn);
    }
  }
}

// ---------------------------------------------------------------------------
extern "C" void kernel_launch(void* const* d_in, const int* in_sizes, int n_in,
                              void* d_out, int out_size, void* d_ws, size_t ws_size,
                              hipStream_t stream) {
  const float* hs = (const float*)d_in[0];
  const float* Wq = (const float*)d_in[1];
  const float* Wk = (const float*)d_in[2];
  const float* Wv = (const float*)d_in[3];
  const float* Wo = (const float*)d_in[4];
  float* out = (float*)d_out;

  const size_t N_HS = (size_t)B_ * S_ * H_;     // 8,388,608
  const size_t N_W  = (size_t)H_ * H_;          // 4,194,304
  const int    M    = B_ * S_;                  // 4096

  char* ws = (char*)d_ws;
  size_t off = 0;
  _Float16* hs16 = (_Float16*)(ws + off); off += N_HS * 2;
  _Float16* wq16 = (_Float16*)(ws + off); off += N_W * 2;
  _Float16* wk16 = (_Float16*)(ws + off); off += N_W * 2;
  _Float16* wv16 = (_Float16*)(ws + off); off += N_W * 2;
  _Float16* wo16 = (_Float16*)(ws + off); off += N_W * 2;
  _Float16* q16  = (_Float16*)(ws + off); off += N_HS * 2;
  _Float16* k16  = (_Float16*)(ws + off); off += N_HS * 2;
  _Float16* v16  = (_Float16*)(ws + off); off += N_HS * 2;
  _Float16* mg16 = (_Float16*)(ws + off); off += N_HS * 2;

  // 1) fp32 -> f16
  {
    int n4 = (int)(N_HS / 4);
    cvt_f32_f16<<<(n4 + 255) / 256, 256, 0, stream>>>(hs, hs16, n4);
    int w4 = (int)(N_W / 4);
    cvt_f32_f16<<<(w4 + 255) / 256, 256, 0, stream>>>(Wq, wq16, w4);
    cvt_f32_f16<<<(w4 + 255) / 256, 256, 0, stream>>>(Wk, wk16, w4);
    cvt_f32_f16<<<(w4 + 255) / 256, 256, 0, stream>>>(Wv, wv16, w4);
    cvt_f32_f16<<<(w4 + 255) / 256, 256, 0, stream>>>(Wo, wo16, w4);
  }

  // 2) QKV projections -> head-split f16 (128x128 tiles)
  dim3 ggrid(M / 128, H_ / 128);
  gemm_xwT<0><<<ggrid, 256, 0, stream>>>(hs16, wq16, (void*)q16, M, H_, H_);
  gemm_xwT<0><<<ggrid, 256, 0, stream>>>(hs16, wk16, (void*)k16, M, H_, H_);
  gemm_xwT<0><<<ggrid, 256, 0, stream>>>(hs16, wv16, (void*)v16, M, H_, H_);

  // 3) attention (256KB dynamic LDS of the WGP's 320KB)
  hipFuncSetAttribute((const void*)attn_kernel,
                      hipFuncAttributeMaxDynamicSharedMemorySize, SM_TOTAL);
  dim3 agrid(S_ / QT_, B_ * NH_);
  attn_kernel<<<agrid, 256, SM_TOTAL, stream>>>(q16, k16, v16, mg16);

  // 4) output projection -> f32
  gemm_xwT<1><<<ggrid, 256, 0, stream>>>(mg16, wo16, (void*)out, M, H_, H_);
}
